// DualEncoderLSTM_64226940945029
// MI455X (gfx1250) — matrix operand order repaired and checked by reading, not compiled, not run on hardware
//
#include <hip/hip_runtime.h>
#include <stdint.h>

// ---------------------------------------------------------------------------
// DualEncoderLSTM on MI455X (gfx1250, wave32, WMMA, async global->LDS).
// - query+reply batches stacked (shared weights) -> effective batch 1024
// - per timestep: Z = [x_t ; h] @ W via v_wmma_f32_16x16x32_bf16, all four
//   gates fused in-register per wave (no Z round trip to HBM)
// - W pre-converted to bf16 *transposed* [n][k] so B tiles are contiguous
//   16B/lane copies -> GLOBAL_LOAD_ASYNC_TO_LDS_B128 (ASYNCcnt) when available
// - K-loop split into embed phase (f32 gather+convert) and hidden phase
//   (pure bf16 async copy) -> no divergent branches in the hot loop
// ---------------------------------------------------------------------------

#define VOCAB   50000
#define EMBED   512
#define HID     1024
#define MAXLEN  100
#define BATCH   512
#define BATCH2  1024
#define KTOT    (EMBED + HID)   /* 1536 */
#define FOURH   (4 * HID)       /* 4096 */
#define FORGET_BIAS 2.0f

typedef __attribute__((ext_vector_type(16))) __bf16   v16bf;
typedef __attribute__((ext_vector_type(8)))  float    v8f;
typedef __attribute__((ext_vector_type(4)))  unsigned v4u;
typedef __attribute__((ext_vector_type(4)))  float    v4f;
typedef __attribute__((ext_vector_type(4)))  int      v4i;

union FragBF { v16bf v; v4u q[2]; };

__device__ __forceinline__ unsigned short f2bf(float f) {
  unsigned u = __float_as_uint(f);
  return (unsigned short)((u + 0x7FFFu + ((u >> 16) & 1u)) >> 16); // RNE
}
__device__ __forceinline__ unsigned pack2(float lo, float hi) {
  return (unsigned)f2bf(lo) | ((unsigned)f2bf(hi) << 16);
}
__device__ __forceinline__ float sigm(float x) {
  return 1.0f / (1.0f + __expf(-x));
}

// ---- async global->LDS (CDNA5), with portable fallback --------------------
// Probe round 2 revealed the builtin's signature: param0 is AS(1) int4*
// (prints as 'int __vector(4) __device__ *'), i.e. (global src, lds dst, ...).
#if defined(__HIP_DEVICE_COMPILE__) && __has_builtin(__builtin_amdgcn_global_load_async_to_lds_b128)
#define HAS_ASYNC_LDS 1
#else
#define HAS_ASYNC_LDS 0
#endif

__device__ __forceinline__ void async_copy16(const unsigned short* __restrict__ g,
                                             unsigned short* l) {
#if HAS_ASYNC_LDS
  __builtin_amdgcn_global_load_async_to_lds_b128(
      (__attribute__((address_space(1))) v4i*)g,
      (__attribute__((address_space(3))) v4i*)l, 0, 0);
#else
  *(v4u*)l = *(const v4u*)g;
#endif
}
__device__ __forceinline__ void async_wait0() {
#if HAS_ASYNC_LDS
#if __has_builtin(__builtin_amdgcn_s_wait_asynccnt)
  __builtin_amdgcn_s_wait_asynccnt(0);
#else
  asm volatile("s_wait_asynccnt 0x0" ::: "memory");
#endif
#endif
}

// ---------------------------------------------------------------------------
__global__ void k_init(float* __restrict__ C, float* __restrict__ Hf,
                       unsigned short* __restrict__ Hbf0,
                       unsigned short* __restrict__ Hbf1,
                       float* __restrict__ lossAcc) {
  int i = blockIdx.x * blockDim.x + threadIdx.x;
  if (i < BATCH2 * HID) { C[i] = 0.f; Hf[i] = 0.f; Hbf0[i] = 0; Hbf1[i] = 0; }
  if (i == 0) *lossAcc = 0.f;
}

// f32 -> bf16, TRANSPOSED: WbfT[n][k] = W[k][n], MbfT[n][k] = M[k][n]
__global__ void k_convert(const float* __restrict__ W, const float* __restrict__ M,
                          unsigned short* __restrict__ WbfT,
                          unsigned short* __restrict__ MbfT) {
  int i = blockIdx.x * blockDim.x + threadIdx.x;
  if (i < KTOT * FOURH) {
    int n = i / KTOT, k = i - n * KTOT;            // coalesced writes
    WbfT[i] = f2bf(W[(size_t)k * FOURH + n]);
  }
  if (i < HID * HID) {
    int n = i / HID, k = i - n * HID;
    MbfT[i] = f2bf(M[(size_t)k * HID + n]);
  }
}

// ---- shared WMMA chunk: A[64x32] x B[512 rows(n) x 32(k)] -> 16 WMMAs -----
__device__ __forceinline__ void wmma_chunk4g(const unsigned short* smA,
                                             const unsigned short* smB,
                                             int mh, int nh, int half, int l16,
                                             v8f acc[4][2][2]) {
  FragBF a[2];
#pragma unroll
  for (int ms = 0; ms < 2; ++ms) {
    int mrow = mh * 32 + ms * 16 + l16;
    a[ms].q[0] = *(const v4u*)&smA[mrow * 32 + half * 8];       // K 0..7 / 8..15
    a[ms].q[1] = *(const v4u*)&smA[mrow * 32 + 16 + half * 8];  // K 16..23 / 24..31
  }
  FragBF b[4][2];
#pragma unroll
  for (int g = 0; g < 4; ++g)
#pragma unroll
    for (int ns = 0; ns < 2; ++ns) {
      int nrow = g * 128 + nh * 32 + ns * 16 + l16;
      b[g][ns].q[0] = *(const v4u*)&smB[nrow * 32 + half * 8];
      b[g][ns].q[1] = *(const v4u*)&smB[nrow * 32 + 16 + half * 8];
    }
#pragma unroll
  for (int g = 0; g < 4; ++g)
#pragma unroll
    for (int ns = 0; ns < 2; ++ns)
#pragma unroll
      for (int ms = 0; ms < 2; ++ms)
        acc[g][ms][ns] = __builtin_amdgcn_wmma_f32_16x16x32_bf16(
            false, a[ms].v, false, b[g][ns].v, (short)0, acc[g][ms][ns], false, false);
}

// B tile copy: 4 gates x 128 n-rows x 32 k (contiguous per row in WbfT)
__device__ __forceinline__ void copyB4g(const unsigned short* __restrict__ WbfT,
                                        unsigned short* smB, int tid, int nBase, int kc) {
#pragma unroll
  for (int it = 0; it < 8; ++it) {
    int idx = it * 256 + tid;
    int nr  = idx >> 2;                              // 0..511
    int seg = idx & 3;
    int col = (nr >> 7) * HID + nBase + (nr & 127);  // gate*1024 + n
    async_copy16(WbfT + (size_t)col * KTOT + kc + seg * 8, &smB[nr * 32 + seg * 8]);
  }
}

// ---------------------------------------------------------------------------
// Fused LSTM step. Grid (16,8), block 256 (8 waves).
// ---------------------------------------------------------------------------
__global__ __launch_bounds__(256)
void k_lstm_step(int t,
                 const float* __restrict__ emb,
                 const unsigned short* __restrict__ WbfT,
                 const float* __restrict__ bias,
                 const float* __restrict__ p_i,
                 const float* __restrict__ p_f,
                 const float* __restrict__ p_o,
                 const int* __restrict__ qtok, const int* __restrict__ rtok,
                 const int* __restrict__ qlen, const int* __restrict__ rlen,
                 const unsigned short* __restrict__ HbfIn,
                 unsigned short* __restrict__ HbfOut,
                 float* __restrict__ Hf, float* __restrict__ Cst)
{
  __shared__ __align__(16) unsigned short smA[64 * 32];     // [m][k]
  __shared__ __align__(16) unsigned short smB[512 * 32];    // [gate*128+n][k]
  __shared__ int sIdx[64];

  const int tid = threadIdx.x, lane = tid & 31, wave = tid >> 5;
  const int mh = wave >> 2, nh = wave & 3, half = lane >> 4, l16 = lane & 15;
  const int blockM = blockIdx.x * 64, nBase = blockIdx.y * 128;
  const int arow = tid >> 2, aseg = tid & 3;

  if (tid < 64) {
    int m = blockM + tid;
    sIdx[tid] = (m < BATCH) ? qtok[m * MAXLEN + t] : rtok[(m - BATCH) * MAXLEN + t];
  }

  v8f acc[4][2][2];
  const v8f vzero = {0.f,0.f,0.f,0.f,0.f,0.f,0.f,0.f};
#pragma unroll
  for (int g = 0; g < 4; ++g)
#pragma unroll
    for (int ms = 0; ms < 2; ++ms)
#pragma unroll
      for (int ns = 0; ns < 2; ++ns) acc[g][ms][ns] = vzero;

  // -------- phase 1: K = embedding columns (gather f32 -> bf16) --------
  for (int kc = 0; kc < EMBED; kc += 32) {
    __syncthreads();
    {
      const v4f* src = (const v4f*)(emb + (size_t)sIdx[arow] * EMBED + kc + aseg * 8);
      v4f f0 = src[0], f1 = src[1];
      v4u av;
      av.x = pack2(f0.x, f0.y); av.y = pack2(f0.z, f0.w);
      av.z = pack2(f1.x, f1.y); av.w = pack2(f1.z, f1.w);
      *(v4u*)&smA[arow * 32 + aseg * 8] = av;
    }
    copyB4g(WbfT, smB, tid, nBase, kc);
    async_wait0();
    __syncthreads();
    wmma_chunk4g(smA, smB, mh, nh, half, l16, acc);
  }

  // -------- phase 2: K = hidden columns (bf16, fully async) --------
  for (int kc = 0; kc < HID; kc += 32) {
    __syncthreads();
    async_copy16(HbfIn + (size_t)(blockM + arow) * HID + kc + aseg * 8,
                 &smA[arow * 32 + aseg * 8]);
    copyB4g(WbfT, smB, tid, nBase, EMBED + kc);
    if (kc + 32 < HID)
      __builtin_prefetch(WbfT + (size_t)(nBase) * KTOT + EMBED + kc + 32, 0, 1);
    async_wait0();
    __syncthreads();
    wmma_chunk4g(smA, smB, mh, nh, half, l16, acc);
  }

  // -------- fused gate epilogue --------
#pragma unroll
  for (int ms = 0; ms < 2; ++ms) {
#pragma unroll
    for (int ns = 0; ns < 2; ++ns) {
      int n = nBase + nh * 32 + ns * 16 + l16;
      float bi = bias[n], bj = bias[HID + n], bfv = bias[2 * HID + n], bo = bias[3 * HID + n];
      float pi = p_i[n], pf = p_f[n], po = p_o[n];
#pragma unroll
      for (int v = 0; v < 8; ++v) {
        int m = blockM + mh * 32 + ms * 16 + half * 8 + v;  // C-layout: M = v + 8*(lane/16)
        size_t off = (size_t)m * HID + n;
        float zi = acc[0][ms][ns][v] + bi;
        float zj = acc[1][ms][ns][v] + bj;
        float zf = acc[2][ms][ns][v] + bfv;
        float zo = acc[3][ms][ns][v] + bo;
        float cold = Cst[off];
        float ig = sigm(zi + pi * cold);
        float fg = sigm(zf + FORGET_BIAS + pf * cold);
        float cn = fg * cold + ig * tanhf(zj);
        float og = sigm(zo + po * cn);
        float hn = og * tanhf(cn);
        int   len   = (m < BATCH) ? qlen[m] : rlen[m - BATCH];
        bool  valid = t < len;
        float cw = valid ? cn : cold;
        float hw = valid ? hn : Hf[off];
        Cst[off] = cw;
        Hf[off]  = hw;
        HbfOut[off] = f2bf(hw);
      }
    }
  }
}

// ---------------------------------------------------------------------------
// Generic single-output GEMM tile body used by qt and dist kernels.
// A: [m][k] rows of Abf, B: [n][k] rows of Brows (both bf16, k contiguous).
// ---------------------------------------------------------------------------
__device__ __forceinline__ void gemm_body(const unsigned short* __restrict__ Abf,
                                          const unsigned short* __restrict__ Brows,
                                          size_t ldA, size_t ldB,
                                          unsigned short* smA, unsigned short* smB,
                                          int tid, int blockM, int nBase,
                                          int mh, int nh, int half, int l16,
                                          v8f acc[2][2]) {
  const int arow = tid >> 2, aseg = tid & 3;
  for (int kc = 0; kc < HID; kc += 32) {
    __syncthreads();
    async_copy16(Abf + (size_t)(blockM + arow) * ldA + kc + aseg * 8,
                 &smA[arow * 32 + aseg * 8]);
#pragma unroll
    for (int it = 0; it < 2; ++it) {
      int idx = it * 256 + tid, nr = idx >> 2, seg = idx & 3;
      async_copy16(Brows + (size_t)(nBase + nr) * ldB + kc + seg * 8,
                   &smB[nr * 32 + seg * 8]);
    }
    async_wait0();
    __syncthreads();
    FragBF a[2];
#pragma unroll
    for (int ms = 0; ms < 2; ++ms) {
      int mrow = mh * 32 + ms * 16 + l16;
      a[ms].q[0] = *(const v4u*)&smA[mrow * 32 + half * 8];
      a[ms].q[1] = *(const v4u*)&smA[mrow * 32 + 16 + half * 8];
    }
#pragma unroll
    for (int ns = 0; ns < 2; ++ns) {
      FragBF b;
      int nrow = nh * 32 + ns * 16 + l16;
      b.q[0] = *(const v4u*)&smB[nrow * 32 + half * 8];
      b.q[1] = *(const v4u*)&smB[nrow * 32 + 16 + half * 8];
#pragma unroll
      for (int ms = 0; ms < 2; ++ms)
        acc[ms][ns] = __builtin_amdgcn_wmma_f32_16x16x32_bf16(
            false, a[ms].v, false, b.v, (short)0, acc[ms][ns], false, false);
    }
  }
}

// qt = q_enc @ M : A = Hbf rows 0..511, B = MbfT rows (already [n][k]).
__global__ __launch_bounds__(256)
void k_gemm_qt(const unsigned short* __restrict__ Hbf,
               const unsigned short* __restrict__ MbfT,
               unsigned short* __restrict__ Qtbf) {
  __shared__ __align__(16) unsigned short smA[64 * 32];
  __shared__ __align__(16) unsigned short smB[128 * 32];
  const int tid = threadIdx.x, lane = tid & 31, wave = tid >> 5;
  const int mh = wave >> 2, nh = wave & 3, half = lane >> 4, l16 = lane & 15;
  const int blockM = blockIdx.x * 64, nBase = blockIdx.y * 128;
  v8f acc[2][2];
  const v8f vzero = {0.f,0.f,0.f,0.f,0.f,0.f,0.f,0.f};
  acc[0][0] = vzero; acc[0][1] = vzero; acc[1][0] = vzero; acc[1][1] = vzero;
  gemm_body(Hbf, MbfT, HID, HID, smA, smB, tid, blockM, nBase, mh, nh, half, l16, acc);
#pragma unroll
  for (int ms = 0; ms < 2; ++ms)
#pragma unroll
    for (int ns = 0; ns < 2; ++ns) {
      int n = nBase + nh * 32 + ns * 16 + l16;
#pragma unroll
      for (int v = 0; v < 8; ++v) {
        int m = blockM + mh * 32 + ms * 16 + half * 8 + v;
        Qtbf[(size_t)m * HID + n] = f2bf(acc[ms][ns][v]);
      }
    }
}

// distances = qt @ r_enc^T : B rows are r_enc rows (row-major = [n][k] already)
__global__ __launch_bounds__(256)
void k_gemm_dist(const unsigned short* __restrict__ Qtbf,
                 const unsigned short* __restrict__ Hbf,   // rows 512.. = r_enc
                 float* __restrict__ dist) {
  __shared__ __align__(16) unsigned short smA[64 * 32];
  __shared__ __align__(16) unsigned short smB[128 * 32];
  const int tid = threadIdx.x, lane = tid & 31, wave = tid >> 5;
  const int mh = wave >> 2, nh = wave & 3, half = lane >> 4, l16 = lane & 15;
  const int blockM = blockIdx.x * 64, nBase = blockIdx.y * 128;
  v8f acc[2][2];
  const v8f vzero = {0.f,0.f,0.f,0.f,0.f,0.f,0.f,0.f};
  acc[0][0] = vzero; acc[0][1] = vzero; acc[1][0] = vzero; acc[1][1] = vzero;
  gemm_body(Qtbf, Hbf + (size_t)BATCH * HID, HID, HID,
            smA, smB, tid, blockM, nBase, mh, nh, half, l16, acc);
#pragma unroll
  for (int ms = 0; ms < 2; ++ms)
#pragma unroll
    for (int ns = 0; ns < 2; ++ns) {
      int n = nBase + nh * 32 + ns * 16 + l16;
#pragma unroll
      for (int v = 0; v < 8; ++v) {
        int m = blockM + mh * 32 + ms * 16 + half * 8 + v;
        dist[(size_t)m * BATCH + n] = acc[ms][ns][v];
      }
    }
}

// ---------------------------------------------------------------------------
// loss = mean(softplus(d) - eye*d)
// ---------------------------------------------------------------------------
__global__ void k_loss_partial(const float* __restrict__ d, float* __restrict__ accp) {
  __shared__ float red[256];
  int tid = threadIdx.x;
  float s = 0.f;
  for (int i = blockIdx.x * 256 + tid; i < BATCH * BATCH; i += gridDim.x * 256) {
    float x  = d[i];
    float sp = fmaxf(x, 0.f) + log1pf(__expf(-fabsf(x)));
    int r = i / BATCH, c = i - r * BATCH;
    s += sp - ((r == c) ? x : 0.f);
  }
  red[tid] = s;
  __syncthreads();
  for (int o = 128; o > 0; o >>= 1) {
    if (tid < o) red[tid] += red[tid + o];
    __syncthreads();
  }
  if (tid == 0) atomicAdd(accp, red[0]);
}

__global__ void k_loss_final(const float* __restrict__ accp, float* __restrict__ out) {
  out[0] = accp[0] * (1.0f / ((float)BATCH * (float)BATCH));
}

// ---------------------------------------------------------------------------
extern "C" void kernel_launch(void* const* d_in, const int* in_sizes, int n_in,
                              void* d_out, int out_size, void* d_ws, size_t ws_size,
                              hipStream_t stream) {
  (void)in_sizes; (void)n_in; (void)out_size; (void)ws_size;
  const float* emb  = (const float*)d_in[0];
  const float* W    = (const float*)d_in[1];
  const float* bias = (const float*)d_in[2];
  const float* p_i  = (const float*)d_in[3];
  const float* p_f  = (const float*)d_in[4];
  const float* p_o  = (const float*)d_in[5];
  const float* M    = (const float*)d_in[6];
  const int* qtok   = (const int*)d_in[7];
  const int* rtok   = (const int*)d_in[8];
  const int* qlen   = (const int*)d_in[9];
  const int* rlen   = (const int*)d_in[10];

  float* dist    = (float*)d_out;
  float* lossOut = dist + BATCH * BATCH;

  char* ws = (char*)d_ws;
  size_t off = 0;
  auto alloc = [&](size_t bytes) -> void* {
    void* p = ws + off;
    off += (bytes + 255) & ~(size_t)255;
    return p;
  };
  unsigned short* WbfT = (unsigned short*)alloc((size_t)FOURH * KTOT * 2);  // 12.6 MB
  unsigned short* MbfT = (unsigned short*)alloc((size_t)HID * HID * 2);
  unsigned short* Hbf0 = (unsigned short*)alloc((size_t)BATCH2 * HID * 2);
  unsigned short* Hbf1 = (unsigned short*)alloc((size_t)BATCH2 * HID * 2);
  float* Hf  = (float*)alloc((size_t)BATCH2 * HID * 4);
  float* Cst = (float*)alloc((size_t)BATCH2 * HID * 4);
  unsigned short* Qtbf = (unsigned short*)alloc((size_t)BATCH * HID * 2);
  float* lossAcc = (float*)alloc(256);

  k_init<<<(BATCH2 * HID) / 256, 256, 0, stream>>>(Cst, Hf, Hbf0, Hbf1, lossAcc);
  k_convert<<<(KTOT * FOURH) / 256, 256, 0, stream>>>(W, M, WbfT, MbfT);

  for (int t = 0; t < MAXLEN; ++t) {
    const unsigned short* hin = (t & 1) ? Hbf1 : Hbf0;
    unsigned short* hout      = (t & 1) ? Hbf0 : Hbf1;
    k_lstm_step<<<dim3(16, 8), 256, 0, stream>>>(
        t, emb, WbfT, bias, p_i, p_f, p_o, qtok, rtok, qlen, rlen,
        hin, hout, Hf, Cst);
  }
  // MAXLEN even -> final hidden state (bf16) lands in Hbf0

  k_gemm_qt<<<dim3(8, 8), 256, 0, stream>>>(Hbf0, MbfT, Qtbf);
  k_gemm_dist<<<dim3(8, 4), 256, 0, stream>>>(Qtbf, Hbf0, dist);
  k_loss_partial<<<256, 256, 0, stream>>>(dist, lossAcc);
  k_loss_final<<<1, 1, 0, stream>>>(lossAcc, lossOut);
}
